// SpikingMambaBlock_8641474199757
// MI455X (gfx1250) — matrix operand order repaired and checked by reading, not compiled, not run on hardware
//
#include <hip/hip_runtime.h>
#include <hip/hip_bf16.h>

// ---------------------------------------------------------------------------
// SpikingMambaBlock on MI455X (gfx1250):
//   bf16 WMMA GEMMs with async-to-LDS double-buffered staging + wave32 scan.
// ---------------------------------------------------------------------------

typedef __attribute__((ext_vector_type(16))) __bf16 v16bf;
typedef __attribute__((ext_vector_type(8)))  __bf16 v8bf;
typedef __attribute__((ext_vector_type(8)))  float  v8f;

#define L_SEQ  2048
#define D_DIM  768
#define DI_DIM 1536
#define NST    16
#define P_DIM  (DI_DIM + 2 * NST)   // 1568
#define TWO_DI (2 * DI_DIM)         // 3072

#define KTILE  64    // K-slab staged per barrier (2 WMMA k-steps)
#define APAD   72    // padded LDS row stride (elements): 144B, conflict-free

// ---------------------------------------------------------------------------
// WMMA helpers (CDNA5 layouts, wave32)
// ---------------------------------------------------------------------------
__device__ __forceinline__ v8f wmma_bf16f32(v16bf a, v16bf b, v8f c) {
  // emits v_wmma_f32_16x16x32_bf16
  return __builtin_amdgcn_wmma_f32_16x16x32_bf16(
      /*neg_a=*/false, a, /*neg_b=*/false, b,
      /*c_mod=*/(short)0, c, /*reuse_a=*/false, /*reuse_b=*/false);
}

// A fragment from an LDS row (32 valid elements at rowp).
// Lane l: row m = l&15, half = l>>4.
// elems j=0..7  -> k = half*8 + j ; elems j=8..15 -> k = 16 + half*8 + j
__device__ __forceinline__ v16bf lds_frag_a(const __bf16* rowp, int half) {
  const __bf16* p0 = rowp + half * 8;
  v8bf lo = *(const v8bf*)p0;
  v8bf hi = *(const v8bf*)(p0 + 16);
  v16bf r;
#pragma unroll
  for (int j = 0; j < 8; ++j) { r[j] = lo[j]; r[j + 8] = hi[j]; }
  return r;
}

// B fragment from an LDS row of transposed weights (row = output col n).
// Lane l: col n = l&15, half = l>>4; elems j=0..7 -> k = half*8 + j ; elems j=8..15 -> k = 16 + half*8 + j
__device__ __forceinline__ v16bf lds_frag_b(const __bf16* rowp, int half) {
  const __bf16* p0 = rowp + half * 8;
  v8bf lo = *(const v8bf*)p0;
  v8bf hi = *(const v8bf*)(p0 + 16);
  v16bf r;
#pragma unroll
  for (int j = 0; j < 8; ++j) { r[j] = lo[j]; r[j + 8] = hi[j]; }
  return r;
}

// Async 16-byte global -> LDS copy (ASYNCcnt-tracked). The LDS address is the
// low 32 bits of the generic pointer (flat->LDS aperture truncation).
__device__ __forceinline__ void async_copy_b128(const void* gsrc, void* ldst) {
  uint32_t laddr = (uint32_t)(uintptr_t)ldst;
  asm volatile("global_load_async_to_lds_b128 %0, %1, off"
               :: "v"(laddr), "v"(gsrc) : "memory");
}
__device__ __forceinline__ void wait_async0() {
  asm volatile("s_wait_asynccnt 0x0" ::: "memory");
}

// ---------------------------------------------------------------------------
// Generic bf16 GEMM:  C(MxN) f32 = A(MxK) bf16 * Bt(NxK)^T bf16 + bias
// Block: 8 waves, 64(M) x 128(N) tile; wave: 32x32 (2x2 WMMA tiles).
// K-slabs of 64 staged in LDS via async loads, double buffered
// (one s_wait_asynccnt + barrier per 8 WMMAs per wave).
// M multiple of 64, K multiple of 64, N multiple of 16 (tile-granular guard).
// ---------------------------------------------------------------------------
__global__ __launch_bounds__(256) void k_gemm_bf16(
    const __bf16* __restrict__ A, const __bf16* __restrict__ Bt,
    const float* __restrict__ bias, float* __restrict__ C,
    int M, int N, int K) {
  __shared__ __align__(16) __bf16 sA[2][64 * APAD];
  __shared__ __align__(16) __bf16 sB[2][128 * APAD];

  const int tid  = threadIdx.x;
  const int wave = tid >> 5;
  const int lane = tid & 31;
  const int half = lane >> 4;
  const int lm   = lane & 15;
  const int mw   = wave >> 2;       // 0..1
  const int nw   = wave & 3;        // 0..3
  const int blockM = blockIdx.y * 64;
  const int blockN = blockIdx.x * 128;
  const int tileM  = blockM + mw * 32;
  const int tileN  = blockN + nw * 32;
  const bool bn0 = (tileN     ) < N;   // wave-uniform guards
  const bool bn1 = (tileN + 16) < N;

  auto issue = [&](int buf, int k0) {
    // A slab: 64 rows x 64 elems = 512 16B-chunks, 2 per thread
#pragma unroll
    for (int i = 0; i < 2; ++i) {
      int c = tid + i * 256;
      int row = c >> 3, seg = c & 7;
      async_copy_b128(A + (size_t)(blockM + row) * K + k0 + seg * 8,
                      &sA[buf][row * APAD + seg * 8]);
    }
    // B slab: 128 rows x 64 elems = 1024 16B-chunks, 4 per thread
#pragma unroll
    for (int i = 0; i < 4; ++i) {
      int c = tid + i * 256;
      int row = c >> 3, seg = c & 7;
      if (blockN + row < N)
        async_copy_b128(Bt + (size_t)(blockN + row) * K + k0 + seg * 8,
                        &sB[buf][row * APAD + seg * 8]);
    }
  };

  v8f acc[2][2] = {};
  issue(0, 0);
  for (int k0 = 0; k0 < K; k0 += KTILE) {
    const int buf = (k0 / KTILE) & 1;
    wait_async0();        // this wave's slab landed in LDS
    __syncthreads();      // everyone's slab landed; prior reads done
    if (k0 + KTILE < K) issue(buf ^ 1, k0 + KTILE);

#pragma unroll
    for (int s = 0; s < 2; ++s) {     // two 32-wide k-steps per slab
      const int ko = s * 32;
      v16bf a0 = lds_frag_a(&sA[buf][(mw * 32 + lm     ) * APAD + ko], half);
      v16bf a1 = lds_frag_a(&sA[buf][(mw * 32 + 16 + lm) * APAD + ko], half);
      if (bn0) {
        v16bf b0 = lds_frag_b(&sB[buf][(nw * 32 + lm) * APAD + ko], half);
        acc[0][0] = wmma_bf16f32(a0, b0, acc[0][0]);
        acc[1][0] = wmma_bf16f32(a1, b0, acc[1][0]);
      }
      if (bn1) {
        v16bf b1 = lds_frag_b(&sB[buf][(nw * 32 + 16 + lm) * APAD + ko], half);
        acc[0][1] = wmma_bf16f32(a0, b1, acc[0][1]);
        acc[1][1] = wmma_bf16f32(a1, b1, acc[1][1]);
      }
    }
  }

#pragma unroll
  for (int mi = 0; mi < 2; ++mi) {
#pragma unroll
    for (int ni = 0; ni < 2; ++ni) {
      int nb = tileN + ni * 16;
      if (nb >= N) continue;
      int n = nb + lm;
      float bv = bias ? bias[n] : 0.0f;
#pragma unroll
      for (int r = 0; r < 8; ++r) {
        int m = tileM + mi * 16 + half * 8 + r;
        C[(size_t)m * N + n] = acc[mi][ni][r] + bv;
      }
    }
  }
}

// ---------------------------------------------------------------------------
// Elementwise / glue kernels
// ---------------------------------------------------------------------------
__global__ void k_f32_to_bf16(const float* __restrict__ in,
                              __bf16* __restrict__ out, size_t n) {
  size_t i = (size_t)blockIdx.x * blockDim.x + threadIdx.x;
  if (i < n) out[i] = (__bf16)in[i];
}

// W: (K x N) f32 row-major  ->  Wt: (N x K) bf16 row-major
__global__ void k_transpose_bf16(const float* __restrict__ W,
                                 __bf16* __restrict__ Wt, int K, int N) {
  size_t i = (size_t)blockIdx.x * blockDim.x + threadIdx.x;
  size_t total = (size_t)K * N;
  if (i >= total) return;
  int n = (int)(i / K);
  int k = (int)(i % K);
  Wt[i] = (__bf16)W[(size_t)k * N + n];
}

// causal depthwise conv (K=4, left pad 3) + LIF spike (vth=1.0)
__global__ void k_conv_spike(const float* __restrict__ xin, int ld,
                             const float* __restrict__ cw,
                             const float* __restrict__ cb,
                             __bf16* __restrict__ xs, int L, int Di) {
  size_t i = (size_t)blockIdx.x * blockDim.x + threadIdx.x;
  if (i >= (size_t)L * Di) return;
  int t = (int)(i / Di), d = (int)(i % Di);
  float acc = cb[d];
#pragma unroll
  for (int k = 0; k < 4; ++k) {
    int tt = t - 3 + k;
    if (tt >= 0) acc += cw[d * 4 + k] * xin[(size_t)tt * ld + d];
  }
  xs[i] = (__bf16)((acc >= 1.0f) ? 1.0f : 0.0f);
}

// split p (L x 1568) -> delta_raw bf16 (L x 1536), Bp (L x 16), Cp (L x 16)
__global__ void k_split_p(const float* __restrict__ p,
                          __bf16* __restrict__ draw,
                          float* __restrict__ Bp, float* __restrict__ Cp, int L) {
  size_t i = (size_t)blockIdx.x * blockDim.x + threadIdx.x;
  if (i >= (size_t)L * P_DIM) return;
  int t = (int)(i / P_DIM), c = (int)(i % P_DIM);
  float v = p[i];
  if (c < DI_DIM)            draw[(size_t)t * DI_DIM + c] = (__bf16)v;
  else if (c < DI_DIM + NST) Bp[t * NST + (c - DI_DIM)] = v;
  else                       Cp[t * NST + (c - DI_DIM - NST)] = v;
}

__global__ void k_softplus(const float* __restrict__ in,
                           float* __restrict__ out, size_t n) {
  size_t i = (size_t)blockIdx.x * blockDim.x + threadIdx.x;
  if (i >= n) return;
  float v = in[i];
  out[i] = (v > 20.0f) ? v : log1pf(expf(v));
}

// Selective scan: one thread per (d, n) state; 16-lane shfl reduce over n.
// Output fused with +D*xs and silu(res) gating, stored bf16 for GEMM4.
__global__ __launch_bounds__(256) void k_scan(
    const float* __restrict__ delta, const __bf16* __restrict__ xs,
    const float* __restrict__ Bp, const float* __restrict__ Cp,
    const float* __restrict__ A_log, const float* __restrict__ Dp,
    const float* __restrict__ res, int ld_res,
    __bf16* __restrict__ ybf, int L, int Di) {
  int gid = blockIdx.x * blockDim.x + threadIdx.x;
  int n = gid & 15;
  int d = gid >> 4;
  if (d >= Di) return;
  const float Aneg = -expf(A_log[d * NST + n]);
  const float dpar = Dp[d];
  float h = 0.0f;
  for (int t = 0; t < L; ++t) {
    float dt  = delta[(size_t)t * Di + d];
    float xsv = (float)xs[(size_t)t * Di + d];
    float a   = expf(Aneg * dt);
    h = a * h + dt * Bp[t * NST + n] * xsv;
    float contrib = h * Cp[t * NST + n];
#pragma unroll
    for (int off = 1; off < 16; off <<= 1)
      contrib += __shfl_xor(contrib, off, 32);
    if (n == 0) {
      float y = contrib + dpar * xsv;
      float r = res[(size_t)t * ld_res + d];
      float s = r / (1.0f + expf(-r));      // silu
      ybf[(size_t)t * Di + d] = (__bf16)(y * s);
    }
  }
}

// residual + LayerNorm + spike; one block per row, D=768 = 256 threads x 3
__global__ __launch_bounds__(256) void k_add_ln_spike(
    const float* __restrict__ x, const float* __restrict__ obuf,
    const float* __restrict__ gamma, const float* __restrict__ beta,
    float* __restrict__ out, int D) {
  int t = blockIdx.x;
  __shared__ float s1[8], s2[8];
  float vals[3];
  float sum = 0.0f, sq = 0.0f;
#pragma unroll
  for (int i = 0; i < 3; ++i) {
    int c = threadIdx.x + i * 256;
    float v = x[(size_t)t * D + c] + obuf[(size_t)t * D + c];
    vals[i] = v; sum += v; sq += v * v;
  }
#pragma unroll
  for (int off = 16; off > 0; off >>= 1) {
    sum += __shfl_xor(sum, off, 32);
    sq  += __shfl_xor(sq,  off, 32);
  }
  int w = threadIdx.x >> 5;
  if ((threadIdx.x & 31) == 0) { s1[w] = sum; s2[w] = sq; }
  __syncthreads();
  sum = 0.0f; sq = 0.0f;
#pragma unroll
  for (int i = 0; i < 8; ++i) { sum += s1[i]; sq += s2[i]; }
  float mean = sum / D;
  float var  = sq / D - mean * mean;
  float rstd = rsqrtf(var + 1e-5f);
#pragma unroll
  for (int i = 0; i < 3; ++i) {
    int c = threadIdx.x + i * 256;
    float u = (vals[i] - mean) * rstd * gamma[c] + beta[c];
    out[(size_t)t * D + c] = (u >= 1.0f) ? 1.0f : 0.0f;
  }
}

// ---------------------------------------------------------------------------
// Orchestration
// ---------------------------------------------------------------------------
extern "C" void kernel_launch(void* const* d_in, const int* in_sizes, int n_in,
                              void* d_out, int out_size, void* d_ws, size_t ws_size,
                              hipStream_t stream) {
  const float* x         = (const float*)d_in[0];
  const float* in_proj_w = (const float*)d_in[1];
  const float* in_proj_b = (const float*)d_in[2];
  const float* conv_w    = (const float*)d_in[3];
  const float* conv_b    = (const float*)d_in[4];
  const float* x_proj_w  = (const float*)d_in[5];
  const float* x_proj_b  = (const float*)d_in[6];
  const float* dt_proj_w = (const float*)d_in[7];
  const float* dt_proj_b = (const float*)d_in[8];
  const float* A_log     = (const float*)d_in[9];
  const float* D_param   = (const float*)d_in[10];
  const float* out_proj_w= (const float*)d_in[11];
  const float* out_proj_b= (const float*)d_in[12];
  const float* ln_gamma  = (const float*)d_in[13];
  const float* ln_beta   = (const float*)d_in[14];
  float* out = (float*)d_out;

  // deterministic bump allocator over d_ws
  char* wsp = (char*)d_ws;
  auto alloc = [&](size_t bytes) -> void* {
    void* p = (void*)wsp;
    wsp += (bytes + 255) & ~(size_t)255;
    return p;
  };

  __bf16* xbf    = (__bf16*)alloc((size_t)L_SEQ * D_DIM * 2);
  __bf16* W1t    = (__bf16*)alloc((size_t)TWO_DI * D_DIM * 2);
  __bf16* W2t    = (__bf16*)alloc((size_t)P_DIM * DI_DIM * 2);
  __bf16* W3t    = (__bf16*)alloc((size_t)DI_DIM * DI_DIM * 2);
  __bf16* W4t    = (__bf16*)alloc((size_t)D_DIM * DI_DIM * 2);
  float*  xr     = (float*) alloc((size_t)L_SEQ * TWO_DI * 4);
  __bf16* xsbf   = (__bf16*)alloc((size_t)L_SEQ * DI_DIM * 2);
  float*  pbuf   = (float*) alloc((size_t)L_SEQ * P_DIM * 4);
  __bf16* drawbf = (__bf16*)alloc((size_t)L_SEQ * DI_DIM * 2);
  float*  Bp     = (float*) alloc((size_t)L_SEQ * NST * 4);
  float*  Cp     = (float*) alloc((size_t)L_SEQ * NST * 4);
  float*  dtbuf  = (float*) alloc((size_t)L_SEQ * DI_DIM * 4);
  float*  delta  = (float*) alloc((size_t)L_SEQ * DI_DIM * 4);
  __bf16* ybf    = (__bf16*)alloc((size_t)L_SEQ * DI_DIM * 2);
  float*  obuf   = (float*) alloc((size_t)L_SEQ * D_DIM * 4);
  (void)ws_size; (void)n_in; (void)in_sizes; (void)out_size;

  auto blocks1d = [](size_t n) { return (unsigned)((n + 255) / 256); };

  // 1) fp32 -> bf16 conversions / weight transposes
  k_f32_to_bf16<<<blocks1d((size_t)L_SEQ * D_DIM), 256, 0, stream>>>(
      x, xbf, (size_t)L_SEQ * D_DIM);
  k_transpose_bf16<<<blocks1d((size_t)D_DIM * TWO_DI), 256, 0, stream>>>(
      in_proj_w, W1t, D_DIM, TWO_DI);
  k_transpose_bf16<<<blocks1d((size_t)DI_DIM * P_DIM), 256, 0, stream>>>(
      x_proj_w, W2t, DI_DIM, P_DIM);
  k_transpose_bf16<<<blocks1d((size_t)DI_DIM * DI_DIM), 256, 0, stream>>>(
      dt_proj_w, W3t, DI_DIM, DI_DIM);
  k_transpose_bf16<<<blocks1d((size_t)DI_DIM * D_DIM), 256, 0, stream>>>(
      out_proj_w, W4t, DI_DIM, D_DIM);

  // 2) GEMM1: xr = x @ in_proj_w + b     (2048 x 3072, K=768)
  {
    dim3 g(TWO_DI / 128, L_SEQ / 64);
    k_gemm_bf16<<<g, 256, 0, stream>>>(xbf, W1t, in_proj_b, xr,
                                       L_SEQ, TWO_DI, D_DIM);
  }
  // 3) causal conv + spike -> xs (bf16)
  k_conv_spike<<<blocks1d((size_t)L_SEQ * DI_DIM), 256, 0, stream>>>(
      xr, TWO_DI, conv_w, conv_b, xsbf, L_SEQ, DI_DIM);

  // 4) GEMM2: p = xs @ x_proj_w + b      (2048 x 1568, K=1536)
  {
    dim3 g((P_DIM + 127) / 128, L_SEQ / 64);
    k_gemm_bf16<<<g, 256, 0, stream>>>(xsbf, W2t, x_proj_b, pbuf,
                                       L_SEQ, P_DIM, DI_DIM);
  }
  // 5) split p -> delta_raw(bf16), Bp, Cp
  k_split_p<<<blocks1d((size_t)L_SEQ * P_DIM), 256, 0, stream>>>(
      pbuf, drawbf, Bp, Cp, L_SEQ);

  // 6) GEMM3: dtbuf = delta_raw @ dt_proj_w + b   (2048 x 1536, K=1536)
  {
    dim3 g(DI_DIM / 128, L_SEQ / 64);
    k_gemm_bf16<<<g, 256, 0, stream>>>(drawbf, W3t, dt_proj_b, dtbuf,
                                       L_SEQ, DI_DIM, DI_DIM);
  }
  // 7) delta = softplus(dtbuf)
  k_softplus<<<blocks1d((size_t)L_SEQ * DI_DIM), 256, 0, stream>>>(
      dtbuf, delta, (size_t)L_SEQ * DI_DIM);

  // 8) selective scan + D*xs + silu(res) gate -> y (bf16)
  {
    dim3 g((DI_DIM * NST) / 256);
    k_scan<<<g, 256, 0, stream>>>(delta, xsbf, Bp, Cp, A_log, D_param,
                                  xr + DI_DIM, TWO_DI, ybf, L_SEQ, DI_DIM);
  }
  // 9) GEMM4: obuf = y @ out_proj_w + b   (2048 x 768, K=1536)
  {
    dim3 g(D_DIM / 128, L_SEQ / 64);
    k_gemm_bf16<<<g, 256, 0, stream>>>(ybf, W4t, out_proj_b, obuf,
                                       L_SEQ, D_DIM, DI_DIM);
  }
  // 10) residual + LayerNorm + spike -> out
  k_add_ln_spike<<<L_SEQ, 256, 0, stream>>>(x, obuf, ln_gamma, ln_beta,
                                            out, D_DIM);
}
